// RelPositionMultiHeadAttention_44822278701358
// MI455X (gfx1250) — matrix operand order repaired and checked by reading, not compiled, not run on hardware
//
#include <hip/hip_runtime.h>
#include <hip/hip_bf16.h>
#include <cstdint>
#include <cstddef>

// Problem constants (match reference)
#define BB   4
#define TT   1024
#define DDim 512
#define HH   8
#define DKK  64
#define PPos 2047            // 2*T - 1
#define NEGF (-3.4028234663852886e38f)

typedef __attribute__((ext_vector_type(16))) __bf16 v16bf;
typedef __attribute__((ext_vector_type(8)))  float  v8f;

union FragBf { v16bf v; uint4 q[2]; };

__device__ __forceinline__ unsigned short f2bf(float f) {
  union { float f; unsigned u; } x; x.f = f;
  unsigned r = x.u + 0x7FFFu + ((x.u >> 16) & 1u);   // round-to-nearest-even
  return (unsigned short)(r >> 16);
}

__device__ __forceinline__ unsigned pack2(float lo, float hi) {
  return (unsigned)f2bf(lo) | ((unsigned)f2bf(hi) << 16);
}

__device__ __forceinline__ v8f wmma_bf16(const FragBf& a, const FragBf& b, v8f c) {
  // (neg_a, A, neg_b, B, c_mod, C, reuse_a, reuse_b) -> v_wmma_f32_16x16x32_bf16
  return __builtin_amdgcn_wmma_f32_16x16x32_bf16(false, a.v, false, b.v, (short)0, c,
                                                 false, false);
}

// ---------------------------------------------------------------------------
// Generic GEMM:  C[M,N] = A[M,K](f32) @ Bm[K,N](f32) + bias[N]
// WG tile 64x128, 8 waves (2x4), each wave 32x32 (2x2 WMMA tiles), K-step 32.
// Staging is branchless + vectorized: A rows out of range are clamped to row
// M-1 (their accumulators only feed guarded stores), loads are float4, and
// bf16 pairs are packed into dwords so LDS stores are b128 (A) / b32 (B).
// Fragment layout used here (contiguous k order, same for A and B):
//   A (16x32): lane(l) m=l%16, h=l/16 holds K in {16h..16h+15}
//   B (32x16): lane(l) n=l%16, h=l/16 holds K in {16h..16h+15}
// ---------------------------------------------------------------------------
__global__ __launch_bounds__(256) void gemm_bias_bf16(
    const float* __restrict__ A, const float* __restrict__ Bm,
    const float* __restrict__ bias, float* __restrict__ C,
    int M, int N, int K)
{
  __shared__ unsigned short Abf[64][32];    // row-major (m, k)
  __shared__ unsigned short Bbf[128][32];   // transposed (n, k): K pairs contiguous

  const int tid  = threadIdx.x;
  const int wave = tid >> 5;
  const int lane = tid & 31;
  const int half = lane >> 4;
  const int ln   = lane & 15;
  const int n0 = blockIdx.x * 128;
  const int m0 = blockIdx.y * 64;
  const int wm = (wave >> 2) * 32;          // 0 / 32
  const int wn = (wave & 3) * 32;           // 0..96

  // staging coordinates (fixed per thread)
  const int ar  = tid >> 2;                 // A row 0..63
  const int ac8 = (tid & 3) * 8;            // A col 0,8,16,24
  int gmA = m0 + ar; if (gmA >= M) gmA = M - 1;     // branchless clamp
  const float* Arow = A + (size_t)gmA * K;

  v8f acc[2][2] = {};

  for (int k0 = 0; k0 < K; k0 += 32) {
    // ---- stage A tile (64x32): two float4 loads -> one b128 LDS store ----
    {
      const float4 f0 = *(const float4*)(Arow + k0 + ac8);
      const float4 f1 = *(const float4*)(Arow + k0 + ac8 + 4);
      uint4 pk;
      pk.x = pack2(f0.x, f0.y);
      pk.y = pack2(f0.z, f0.w);
      pk.z = pack2(f1.x, f1.y);
      pk.w = pack2(f1.z, f1.w);
      *(uint4*)&Abf[ar][ac8] = pk;
      if (k0 + 32 < K) __builtin_prefetch(Arow + k0 + 32 + ac8, 0, 1);
    }
    // ---- stage B tile (32x128) transposed: thread does 2(k) x 4(n) ----
    for (int i = 0; i < 2; ++i) {
      int id = tid + 256 * i;               // 0..511
      int k  = (id >> 5) * 2;               // 0,2,..,30
      int n  = (id & 31) * 4;               // 0,4,..,124
      const float* Bp = Bm + (size_t)(k0 + k) * N + n0 + n;
      const float4 r0 = *(const float4*)(Bp);
      const float4 r1 = *(const float4*)(Bp + N);
      *(unsigned*)&Bbf[n + 0][k] = pack2(r0.x, r1.x);
      *(unsigned*)&Bbf[n + 1][k] = pack2(r0.y, r1.y);
      *(unsigned*)&Bbf[n + 2][k] = pack2(r0.z, r1.z);
      *(unsigned*)&Bbf[n + 3][k] = pack2(r0.w, r1.w);
      if (k0 + 32 < K) __builtin_prefetch(Bp + (size_t)32 * N, 0, 1);
    }
    __syncthreads();

    FragBf fa[2], fb[2];
    for (int i = 0; i < 2; ++i) {
      const unsigned short* row = &Abf[wm + 16 * i + ln][0];
      fa[i].q[0] = *(const uint4*)(row + 16 * half);
      fa[i].q[1] = *(const uint4*)(row + 16 * half + 8);
    }
    for (int j = 0; j < 2; ++j) {
      const unsigned short* row = &Bbf[wn + 16 * j + ln][0];
      fb[j].q[0] = *(const uint4*)(row + 16 * half);
      fb[j].q[1] = *(const uint4*)(row + 16 * half + 8);
    }
    for (int i = 0; i < 2; ++i)
      for (int j = 0; j < 2; ++j)
        acc[i][j] = wmma_bf16(fa[i], fb[j], acc[i][j]);
    __syncthreads();
  }

  // epilogue: C-layout VGPR r -> M = r + 8*half, N = ln
  for (int i = 0; i < 2; ++i)
    for (int j = 0; j < 2; ++j)
      for (int r = 0; r < 8; ++r) {
        int gm = m0 + wm + 16 * i + r + 8 * half;
        int gn = n0 + wn + 16 * j + ln;
        if (gm < M) {
          float v = acc[i][j][r];
          if (bias) v += bias[gn];
          C[(size_t)gm * N + gn] = v;
        }
      }
}

// ---------------------------------------------------------------------------
// Rearrange + bias + f32->bf16:
//   qu/qv/kk : (b,h,t,dk) bf16 ; vT : (b,h,dk,t) bf16 ; pp : (h,p,dk) bf16
// Scattered vT b16 stores are absorbed by the 192MB L2 (whole tensor is 4MB).
// ---------------------------------------------------------------------------
__global__ __launch_bounds__(256) void rearrange_kernel(
    const float* __restrict__ qf, const float* __restrict__ kf,
    const float* __restrict__ vf, const float* __restrict__ pf,
    const float* __restrict__ pbu, const float* __restrict__ pbv,
    unsigned short* __restrict__ qu, unsigned short* __restrict__ qv,
    unsigned short* __restrict__ kk, unsigned short* __restrict__ vT,
    unsigned short* __restrict__ pp)
{
  const long long NTD = (long long)BB * TT * DDim;        // 2,097,152
  long long e = (long long)blockIdx.x * 256 + threadIdx.x;
  if (e < NTD) {
    int d = (int)(e % DDim);
    long long bt = e / DDim;
    int t = (int)(bt % TT);
    int b = (int)(bt / TT);
    int h = d >> 6, dk = d & 63;
    size_t dst = (((size_t)b * HH + h) * TT + t) * DKK + dk;
    float q = qf[e];
    qu[dst] = f2bf(q + pbu[d]);
    qv[dst] = f2bf(q + pbv[d]);
    kk[dst] = f2bf(kf[e]);
    vT[(((size_t)b * HH + h) * DKK + dk) * TT + t] = f2bf(vf[e]);
  } else {
    long long e2 = e - NTD;
    if (e2 < (long long)PPos * DDim) {
      int d  = (int)(e2 % DDim);
      int pi = (int)(e2 / DDim);
      int h = d >> 6, dk = d & 63;
      pp[((size_t)h * PPos + pi) * DKK + dk] = f2bf(pf[e2]);
    }
  }
}

// ---------------------------------------------------------------------------
// Fused rel-pos attention. One WG per (b, h, 16 query rows); 128 threads.
// rel_shift identity: bd[q,k] = q_v[q] . p[T-1+q-k]  (band of 31 p-rows/tile)
// Dynamic LDS layout (floats):
//   scores [16][1024] | probs(bf16) [16][1024] | G [4][16][32]
//   | parts [4][16][64] | red [16][8] | rowmax[16] | rowinv[16]
// ---------------------------------------------------------------------------
__global__ __launch_bounds__(128) void attn_kernel(
    const unsigned short* __restrict__ qu, const unsigned short* __restrict__ qv,
    const unsigned short* __restrict__ kk, const unsigned short* __restrict__ vT,
    const unsigned short* __restrict__ pp, const int* __restrict__ mask,
    float* __restrict__ x)
{
  extern __shared__ float smem[];
  float*          scores = smem;                                  // 16384 f
  unsigned short* probs  = (unsigned short*)(smem + 16 * 1024);   // 16384 bf16
  float*          gbuf   = smem + 16 * 1024 + 8 * 1024;           // 2048 f
  float*          parts  = gbuf + 4 * 16 * 32;                    // 4096 f
  float*          red    = parts + 4 * 16 * 64;                   // 128 f
  float*          rowmax = red + 16 * 8;                          // 16 f
  float*          rowinv = rowmax + 16;                           // 16 f

  const int tid  = threadIdx.x;
  const int wave = tid >> 5;
  const int lane = tid & 31;
  const int half = lane >> 4;
  const int ln   = lane & 15;

  int idx = blockIdx.x;
  const int qt = idx & 63;  idx >>= 6;     // T/16 = 64
  const int h  = idx & 7;
  const int b  = idx >> 3;
  const int q0 = qt * 16;

  const size_t bh = (size_t)b * HH + h;
  const unsigned short* quB = qu + bh * TT * DKK;
  const unsigned short* qvB = qv + bh * TT * DKK;
  const unsigned short* kkB = kk + bh * TT * DKK;
  const unsigned short* vTB = vT + bh * DKK * TT;
  const unsigned short* ppB = pp + (size_t)h * PPos * DKK;

  // A-fragments for this q-tile (invariant over k): K-blocks 0..31 and 32..63
  FragBf aqu[2], aqv[2];
  {
    const unsigned short* ru = quB + (size_t)(q0 + ln) * DKK;
    const unsigned short* rv = qvB + (size_t)(q0 + ln) * DKK;
    for (int kb = 0; kb < 2; ++kb) {
      int K0 = 32 * kb;
      aqu[kb].q[0] = *(const uint4*)(ru + K0 + 16 * half);
      aqu[kb].q[1] = *(const uint4*)(ru + K0 + 16 * half + 8);
      aqv[kb].q[0] = *(const uint4*)(rv + K0 + 16 * half);
      aqv[kb].q[1] = *(const uint4*)(rv + K0 + 16 * half + 8);
    }
  }

  // ---------------- pass 1: scores = (ac + bd) * 1/sqrt(DK) ----------------
  for (int kt = wave; kt < 64; kt += 4) {
    const int k0 = kt * 16;

    // ac = q_u . k^T   (B column n = token k0+n, K = dk)
    FragBf bk0, bk1;
    const unsigned short* krow = kkB + (size_t)(k0 + ln) * DKK;
    bk0.q[0] = *(const uint4*)(krow + 16 * half);
    bk0.q[1] = *(const uint4*)(krow + 16 * half + 8);
    bk1.q[0] = *(const uint4*)(krow + 32 + 16 * half);
    bk1.q[1] = *(const uint4*)(krow + 32 + 16 * half + 8);
    v8f cac = {};
    cac = wmma_bf16(aqu[0], bk0, cac);
    cac = wmma_bf16(aqu[1], bk1, cac);

    // bd band: p rows [c0-15, c0+15], c0 = T-1+q0-k0  -> two 16-col tiles
    const int c0 = TT - 1 + q0 - k0;
    int r0 = c0 - 15 + ln;                       // always >= 0
    int r1 = c0 + 1 + ln; if (r1 > PPos - 1) r1 = PPos - 1;   // row c0+16 unused
    const unsigned short* p0 = ppB + (size_t)r0 * DKK;
    const unsigned short* p1 = ppB + (size_t)r1 * DKK;
    FragBf bp00, bp01, bp10, bp11;
    bp00.q[0] = *(const uint4*)(p0 + 16 * half);
    bp00.q[1] = *(const uint4*)(p0 + 16 * half + 8);
    bp01.q[0] = *(const uint4*)(p0 + 32 + 16 * half);
    bp01.q[1] = *(const uint4*)(p0 + 32 + 16 * half + 8);
    bp10.q[0] = *(const uint4*)(p1 + 16 * half);
    bp10.q[1] = *(const uint4*)(p1 + 16 * half + 8);
    bp11.q[0] = *(const uint4*)(p1 + 32 + 16 * half);
    bp11.q[1] = *(const uint4*)(p1 + 32 + 16 * half + 8);
    v8f g0 = {}, g1 = {};
    g0 = wmma_bf16(aqv[0], bp00, g0);
    g0 = wmma_bf16(aqv[1], bp01, g0);
    g1 = wmma_bf16(aqv[0], bp10, g1);
    g1 = wmma_bf16(aqv[1], bp11, g1);

    // prefetch next k-tile / p-band rows for this wave (GL2 hint)
    if (kt + 4 < 64) {
      __builtin_prefetch(krow + 64 * DKK, 0, 1);
      int rp = c0 - 64 - 15 + ln; if (rp < 0) rp = 0;
      __builtin_prefetch(ppB + (size_t)rp * DKK, 0, 1);
    }

    // park 16x32 band result in per-wave LDS, gather diagonal G[m][15+m-n]
    float* G = gbuf + wave * 16 * 32;
    for (int r = 0; r < 8; ++r) {
      int m = r + 8 * half;
      G[m * 32 + ln]      = g0[r];
      G[m * 32 + 16 + ln] = g1[r];
    }
    for (int r = 0; r < 8; ++r) {
      int m = r + 8 * half;
      float bd = G[m * 32 + 15 + m - ln];
      scores[m * 1024 + k0 + ln] = (cac[r] + bd) * 0.125f;   // 1/sqrt(64)
    }
  }
  __syncthreads();

  // ------------- softmax (8 threads per row), mask applied here ------------
  {
    const int row = tid >> 3, sub = tid & 7;
    const int* mrow = mask + ((size_t)b * TT + q0 + row) * TT;
    float mx = NEGF;
    for (int c = sub; c < 1024; c += 8) {
      float sc = scores[row * 1024 + c];
      if (mrow[c] != 0) { sc = NEGF; scores[row * 1024 + c] = NEGF; }
      mx = fmaxf(mx, sc);
    }
    red[row * 8 + sub] = mx;
    __syncthreads();
    if (sub == 0) {
      float m2 = red[row * 8];
      for (int i = 1; i < 8; ++i) m2 = fmaxf(m2, red[row * 8 + i]);
      rowmax[row] = m2;
    }
    __syncthreads();
    const float rmax = rowmax[row];
    float sum = 0.0f;
    for (int c = sub; c < 1024; c += 8) {
      float sc = scores[row * 1024 + c];
      float ev = (sc < -1e37f) ? 0.0f : __expf(sc - rmax);  // masked -> 0
      probs[row * 1024 + c] = f2bf(ev);                     // unnormalized
      sum += ev;
    }
    __syncthreads();
    red[row * 8 + sub] = sum;
    __syncthreads();
    if (sub == 0) {
      float s2 = 0.0f;
      for (int i = 0; i < 8; ++i) s2 += red[row * 8 + i];
      rowinv[row] = (s2 > 0.0f) ? 1.0f / s2 : 0.0f;         // all-masked row -> 0
    }
    __syncthreads();
  }

  // ---------------- pass 2: O = probs . v  ---------------------------------
  v8f o[4] = {};
  for (int kb = wave; kb < 32; kb += 4) {
    const int K0 = kb * 32;
    FragBf ap;
    const unsigned short* prow = probs + ln * 1024;
    ap.q[0] = *(const uint4*)(prow + K0 + 16 * half);
    ap.q[1] = *(const uint4*)(prow + K0 + 16 * half + 8);
    for (int nt = 0; nt < 4; ++nt) {
      const unsigned short* vrow = vTB + (size_t)(nt * 16 + ln) * TT;
      FragBf fv;
      fv.q[0] = *(const uint4*)(vrow + K0 + 16 * half);
      fv.q[1] = *(const uint4*)(vrow + K0 + 16 * half + 8);
      o[nt] = wmma_bf16(ap, fv, o[nt]);
    }
  }
  {
    float* my = parts + wave * 16 * 64;
    for (int nt = 0; nt < 4; ++nt)
      for (int r = 0; r < 8; ++r)
        my[(r + 8 * half) * 64 + nt * 16 + ln] = o[nt][r];
  }
  __syncthreads();
  for (int i = 0; i < 8; ++i) {
    int e = tid + 128 * i;                 // 0..1023 = m*64 + dk
    int m = e >> 6, dk = e & 63;
    float s = parts[e] + parts[1024 + e] + parts[2048 + e] + parts[3072 + e];
    s *= rowinv[m];
    x[((size_t)b * TT + q0 + m) * DDim + h * 64 + dk] = s;
  }
}

// ---------------------------------------------------------------------------
extern "C" void kernel_launch(void* const* d_in, const int* in_sizes, int n_in,
                              void* d_out, int out_size, void* d_ws, size_t ws_size,
                              hipStream_t stream) {
  (void)in_sizes; (void)n_in; (void)out_size; (void)ws_size;
  const float* query   = (const float*)d_in[0];
  const float* key     = (const float*)d_in[1];
  const float* value   = (const float*)d_in[2];
  const int*   mask    = (const int*)  d_in[3];
  const float* pos_emb = (const float*)d_in[4];
  const float* Wq   = (const float*)d_in[5];
  const float* bq   = (const float*)d_in[6];
  const float* Wk   = (const float*)d_in[7];
  const float* bk   = (const float*)d_in[8];
  const float* Wv   = (const float*)d_in[9];
  const float* bv   = (const float*)d_in[10];
  const float* Wpos = (const float*)d_in[11];
  const float* Wout = (const float*)d_in[12];
  const float* bout = (const float*)d_in[13];
  const float* pbu  = (const float*)d_in[14];
  const float* pbv  = (const float*)d_in[15];
  float* out = (float*)d_out;

  // workspace carve (256B aligned blocks)
  char* p = (char*)d_ws;
  auto carve = [&](size_t bytes) {
    char* r = p; p += (bytes + 255) & ~(size_t)255; return r;
  };
  const size_t FBTD = (size_t)BB * TT * DDim * sizeof(float);
  float* qf = (float*)carve(FBTD);
  float* kf = (float*)carve(FBTD);
  float* vf = (float*)carve(FBTD);
  float* pf = (float*)carve((size_t)PPos * DDim * sizeof(float));
  float* xf = (float*)carve(FBTD);
  const size_t HBTD = (size_t)BB * HH * TT * DKK * sizeof(unsigned short);
  unsigned short* quB = (unsigned short*)carve(HBTD);
  unsigned short* qvB = (unsigned short*)carve(HBTD);
  unsigned short* kkB = (unsigned short*)carve(HBTD);
  unsigned short* vTB = (unsigned short*)carve(HBTD);
  unsigned short* ppB = (unsigned short*)carve((size_t)HH * PPos * DKK * sizeof(unsigned short));

  // 1-4: projections (bf16 WMMA GEMMs, f32 out)
  gemm_bias_bf16<<<dim3(DDim / 128, (BB * TT) / 64), 256, 0, stream>>>(
      query, Wq, bq, qf, BB * TT, DDim, DDim);
  gemm_bias_bf16<<<dim3(DDim / 128, (BB * TT) / 64), 256, 0, stream>>>(
      key, Wk, bk, kf, BB * TT, DDim, DDim);
  gemm_bias_bf16<<<dim3(DDim / 128, (BB * TT) / 64), 256, 0, stream>>>(
      value, Wv, bv, vf, BB * TT, DDim, DDim);
  gemm_bias_bf16<<<dim3(DDim / 128, (PPos + 63) / 64), 256, 0, stream>>>(
      pos_emb, Wpos, nullptr, pf, PPos, DDim, DDim);

  // 5: rearrange + pos biases + bf16 conversion
  {
    long long total = (long long)BB * TT * DDim + (long long)PPos * DDim;
    int blocks = (int)((total + 255) / 256);
    rearrange_kernel<<<blocks, 256, 0, stream>>>(qf, kf, vf, pf, pbu, pbv,
                                                 quB, qvB, kkB, vTB, ppB);
  }

  // 6: fused rel-pos attention
  {
    size_t smem = (size_t)(16 * 1024 + 8 * 1024 + 4 * 16 * 32 + 4 * 16 * 64 +
                           16 * 8 + 16 + 16) * sizeof(float);   // ~120.6 KB
    attn_kernel<<<dim3(BB * HH * (TT / 16)), 128, smem, stream>>>(
        quB, qvB, kkB, vTB, ppB, mask, xf);
  }

  // 7: output projection
  gemm_bias_bf16<<<dim3(DDim / 128, (BB * TT) / 64), 256, 0, stream>>>(
      xf, Wout, bout, out, BB * TT, DDim, DDim);
}
